// VQVAE_19610820673827
// MI455X (gfx1250) — matrix operand order repaired and checked by reading, not compiled, not run on hardware
//
#include <hip/hip_runtime.h>
#include <math.h>

// ---------------------------------------------------------------------------
// VQ-VAE forward for MI455X (gfx1250, wave32).
// All channel-mixing convs, the transposed conv dec_t1 (split by output
// parity into two K=2 convs), and the VQ distance GEMM run on
// v_wmma_f32_16x16x32_f16 (f16 inputs, f32 accumulate).
// LDS holds pre-im2col'd f16 tiles so each WMMA fragment is two aligned
// 16-byte LDS loads.
// ---------------------------------------------------------------------------

typedef __attribute__((ext_vector_type(16))) _Float16 v16h;
typedef __attribute__((ext_vector_type(8)))  _Float16 v8h;
typedef __attribute__((ext_vector_type(8)))  float    v8f;

#define BATCH    64
#define TLEN     8192
#define L1       4096      // after enc conv1 (stride 2)
#define L2       2048      // after enc conv2 (stride 2)
#define HCH      128
#define RCH      32
#define DCH      64
#define KCODES   512
#define NTOK     (BATCH * L2)          // 131072 tokens
#define EPSV     1e-5f

// ---------------------------------------------------------------------------
// init: zero the histogram
// ---------------------------------------------------------------------------
__global__ void k_init(int* __restrict__ hist) {
    int i = blockIdx.x * blockDim.x + threadIdx.x;
    if (i < KCODES) hist[i] = 0;
}

// ---------------------------------------------------------------------------
// RevIN statistics: per-batch mean / std over T=8192 (population variance)
// ---------------------------------------------------------------------------
__global__ void k_revin(const float* __restrict__ x,
                        float* __restrict__ means, float* __restrict__ stds) {
    __shared__ float ssum[256], ssq[256];
    int b = blockIdx.x;
    const float* xb = x + (size_t)b * TLEN;
    float s = 0.f, q = 0.f;
    for (int i = threadIdx.x; i < TLEN; i += 256) {
        float v = xb[i];
        s += v; q += v * v;
    }
    ssum[threadIdx.x] = s; ssq[threadIdx.x] = q;
    __syncthreads();
    for (int off = 128; off > 0; off >>= 1) {
        if (threadIdx.x < off) {
            ssum[threadIdx.x] += ssum[threadIdx.x + off];
            ssq[threadIdx.x]  += ssq[threadIdx.x + off];
        }
        __syncthreads();
    }
    if (threadIdx.x == 0) {
        float mean = ssum[0] / (float)TLEN;
        float var  = ssq[0] / (float)TLEN - mean * mean;
        means[b] = mean;
        stds[b]  = sqrtf(fmaxf(var, 0.f) + EPSV);
    }
}

// ---------------------------------------------------------------------------
// weight prep: f32 -> f16 flat copy (layout [CO][CI*K] is already contiguous)
// ---------------------------------------------------------------------------
__global__ void k_wprep(const float* __restrict__ w, _Float16* __restrict__ o, int n) {
    int i = blockIdx.x * blockDim.x + threadIdx.x;
    if (i < n) o[i] = (_Float16)w[i];
}

// ---------------------------------------------------------------------------
// dec_t1 weight regather: [CI=128][CO=64][K=4] torch-transpose layout ->
// two parity weight sets, each [CO][CI*2] f16:
//   even outputs (p=0, PAD=1): k2=0 -> w[.,.,3], k2=1 -> w[.,.,1]
//   odd  outputs (p=1, PAD=0): k2=0 -> w[.,.,2], k2=1 -> w[.,.,0]
// ---------------------------------------------------------------------------
__global__ void k_tw1prep(const float* __restrict__ w, _Float16* __restrict__ o) {
    int i = blockIdx.x * blockDim.x + threadIdx.x;     // 2 * 64 * 256
    if (i >= 2 * 64 * 256) return;
    int p   = i >> 14;
    int rem = i & 16383;
    int co  = rem >> 8;
    int q   = rem & 255;
    int ci  = q >> 1;
    int k2  = q & 1;
    int k   = (p == 0) ? (k2 == 0 ? 3 : 1) : (k2 == 0 ? 2 : 0);
    o[i] = (_Float16)w[(ci * 64 + co) * 4 + k];
}

// ---------------------------------------------------------------------------
// encoder conv1: 1 -> 64 ch, K=4, stride 2, pad 1, relu; normalizes on the fly
// (reduction dim is only 4 -> not worth WMMA)
// ---------------------------------------------------------------------------
__global__ void k_enc1(const float* __restrict__ x,
                       const float* __restrict__ w, const float* __restrict__ bias,
                       const float* __restrict__ means, const float* __restrict__ stds,
                       float* __restrict__ out) {
    int idx = blockIdx.x * blockDim.x + threadIdx.x;     // B * 64 * L1
    int t = idx & (L1 - 1);
    int o = (idx >> 12) & 63;
    int b = idx >> 18;
    float mean = means[b];
    float inv  = 1.f / stds[b];
    const float* xb = x + (size_t)b * TLEN;
    float acc = bias[o];
    #pragma unroll
    for (int k = 0; k < 4; ++k) {
        int pos = 2 * t - 1 + k;
        float xv = (pos >= 0 && pos < TLEN) ? (xb[pos] - mean) * inv : 0.f;
        acc = fmaf(w[o * 4 + k], xv, acc);
    }
    out[idx] = fmaxf(acc, 0.f);
}

// ---------------------------------------------------------------------------
// WMMA implicit-GEMM conv1d over [B, CI, Lin] -> [B, CO, Lin/STRIDE].
// Block = 8 waves. Input tile staged into LDS already im2col'd as f16 with
// layout [t][CI*K] (+8-half row pad: keeps staging stores spread across
// banks and fragment rows 16B aligned). Each wave computes a 16(co) x 16(t)
// f32 tile: per 32-chunk, A/B fragments are two aligned v8h (b128) loads,
// then one v_wmma_f32_16x16x32_f16. Bias/residual-acc/relu fused at store.
// OSTRIDE=2 supports parity-interleaved ConvTranspose output.
// ---------------------------------------------------------------------------
template<int CI, int CO, int K, int PAD, int STRIDE, int OSTRIDE,
         bool RELU_IN, bool HAS_BIAS, bool ACC_OUT, bool RELU_OUT>
__global__ __launch_bounds__(256)
void k_conv_wmma(const float* __restrict__ in, const _Float16* __restrict__ w16,
                 const float* __restrict__ bias, float* __restrict__ out, int Lin) {
    constexpr int COT  = CO / 16;                 // co tiles per block
    constexpr int WT   = 8 / COT;                 // t sub-tiles per block
    constexpr int TT   = 16 * WT;                 // output positions per block
    constexpr int CIK  = CI * K;                  // reduction dim (multiple of 32)
    constexpr int CIKP = CIK + 8;                 // padded LDS row (16B pad)
    __shared__ __align__(16) _Float16 s[TT * CIKP];

    const int Lout = Lin / STRIDE;
    const int bpr  = Lout / TT;
    const int b    = blockIdx.x / bpr;
    const int t0   = (blockIdx.x % bpr) * TT;     // output tile start
    const float* inb = in + (size_t)b * CI * Lin;

    // stage im2col tile: iterate (ci, k, tpos) tpos-fastest for coalescing
    for (int i = threadIdx.x; i < CIK * TT; i += 256) {
        int ci = i / (K * TT);
        int r  = i - ci * (K * TT);
        int k  = r / TT;
        int tp = r - k * TT;
        int t  = STRIDE * (t0 + tp) - PAD + k;
        float v = (t >= 0 && t < Lin) ? inb[ci * Lin + t] : 0.f;
        if (RELU_IN) v = fmaxf(v, 0.f);
        s[tp * CIKP + ci * K + k] = (_Float16)v;
    }
    __syncthreads();

    const int wave = threadIdx.x >> 5;
    const int lane = threadIdx.x & 31;
    const int co16 = (wave % COT) * 16;
    const int tl   = (wave / COT) * 16;
    const int m    = lane & 15;
    const int half = lane >> 4;
    const int kb   = 16 * half;

    const _Float16* srow = s   + (tl + m) * CIKP;
    const _Float16* wrow = w16 + (size_t)(co16 + m) * CIK;

    v8f acc = {};
    for (int cb = 0; cb < CIK; cb += 32) {
        // A fragment per ISA layout: a[0..7] <-> k = 8*half+j ; a[8..15] <-> k = 16+8*half+j
        v8h a0 = *(const v8h*)(wrow + cb + 8 * half);
        v8h a1 = *(const v8h*)(wrow + cb + 16 + 8 * half);
        // B fragment: bf[0..7] <-> k = 8*half+j ; bf[8..15] <-> k = 16+8*half+j
        v8h b0 = *(const v8h*)(srow + cb + 8 * half);
        v8h b1 = *(const v8h*)(srow + cb + 16 + 8 * half);
        v16h a, bf;
        #pragma unroll
        for (int j = 0; j < 8; ++j) {
            a[j] = a0[j];  a[8 + j] = a1[j];
            bf[j] = b0[j]; bf[8 + j] = b1[j];
        }
        acc = __builtin_amdgcn_wmma_f32_16x16x32_f16(false, a, false, bf,
                                                     (short)0, acc, false, false);
    }

    #pragma unroll
    for (int r = 0; r < 8; ++r) {
        int co = co16 + r + 8 * half;
        float v = acc[r];
        if (HAS_BIAS) v += bias[co];
        float* op = out + ((size_t)b * CO + co) * (Lout * OSTRIDE)
                        + OSTRIDE * (t0 + tl + m);
        if (ACC_OUT) v += *op;
        if (RELU_OUT) v = fmaxf(v, 0.f);
        *op = v;
    }
}

// ---------------------------------------------------------------------------
// VQ prep: f32 codebook -> f16 copy + per-code squared norms
// ---------------------------------------------------------------------------
__global__ void k_vq_prep(const float* __restrict__ cb,
                          _Float16* __restrict__ cb16, float* __restrict__ cnorm) {
    int c = blockIdx.x * blockDim.x + threadIdx.x;
    if (c >= KCODES) return;
    float s = 0.f;
    #pragma unroll 8
    for (int d = 0; d < DCH; ++d) {
        float v = cb[c * DCH + d];
        s += v * v;
        cb16[c * DCH + d] = (_Float16)v;
    }
    cnorm[c] = s;
}

// ---------------------------------------------------------------------------
// VQ main: one wave per 16 tokens. Distances via v_wmma_f32_16x16x32_f16
// (d2 = ||c||^2 - 2 z.c ; ||z||^2 constant per row, irrelevant for argmin).
// ---------------------------------------------------------------------------
__global__ __launch_bounds__(32)
void k_vq(const float* __restrict__ z, const _Float16* __restrict__ cb16,
          const float* __restrict__ cnorm, const float* __restrict__ cb32,
          float* __restrict__ quant, float* __restrict__ idx_out,
          int* __restrict__ hist, float* __restrict__ psse) {
    __shared__ int s_idx[16];
    const int lane = threadIdx.x;
    const int n0   = blockIdx.x * 16;
    const int b    = n0 >> 11;
    const int t0   = n0 & (L2 - 1);
    const int m    = lane & 15;
    const int half = lane >> 4;
    const float* zb = z + (size_t)b * DCH * L2 + t0;

    v16h a0, a1;
    #pragma unroll
    for (int j = 0; j < 8; ++j) {
        int d = 8 * half + ((j < 4) ? 2 * j : 16 + 2 * (j - 4));
        a0[2 * j]     = (_Float16)zb[(d)      * L2 + m];
        a0[2 * j + 1] = (_Float16)zb[(d + 1)  * L2 + m];
        a1[2 * j]     = (_Float16)zb[(d + 32) * L2 + m];
        a1[2 * j + 1] = (_Float16)zb[(d + 33) * L2 + m];
    }

    float bmin[8];
    int   bidx[8];
    #pragma unroll
    for (int r = 0; r < 8; ++r) { bmin[r] = 3.4e38f; bidx[r] = 0; }

    for (int g = 0; g < KCODES / 16; ++g) {
        int code = g * 16 + m;
        const _Float16* cr = cb16 + code * DCH + 8 * half;
        v8h c0 = *(const v8h*)(cr);
        v8h c1 = *(const v8h*)(cr + 16);
        v8h c2 = *(const v8h*)(cr + 32);
        v8h c3 = *(const v8h*)(cr + 48);
        v16h b0, b1;
        #pragma unroll
        for (int j = 0; j < 8; ++j) {
            b0[j] = c0[j]; b0[8 + j] = c1[j];
            b1[j] = c2[j]; b1[8 + j] = c3[j];
        }

        v8f acc = {};
        acc = __builtin_amdgcn_wmma_f32_16x16x32_f16(false, a0, false, b0,
                                                     (short)0, acc, false, false);
        acc = __builtin_amdgcn_wmma_f32_16x16x32_f16(false, a1, false, b1,
                                                     (short)0, acc, false, false);
        float cn = cnorm[code];
        #pragma unroll
        for (int r = 0; r < 8; ++r) {
            float d2 = cn - 2.0f * acc[r];
            if (d2 < bmin[r]) { bmin[r] = d2; bidx[r] = code; }
        }
    }

    #pragma unroll
    for (int off = 1; off < 16; off <<= 1) {
        #pragma unroll
        for (int r = 0; r < 8; ++r) {
            float om = __shfl_xor(bmin[r], off, 32);
            int   oi = __shfl_xor(bidx[r], off, 32);
            if (om < bmin[r] || (om == bmin[r] && oi < bidx[r])) {
                bmin[r] = om; bidx[r] = oi;
            }
        }
    }
    if (m == 0) {
        #pragma unroll
        for (int r = 0; r < 8; ++r) s_idx[half * 8 + r] = bidx[r];
    }
    __syncthreads();

    if (lane < 16) {
        int code = s_idx[lane];
        idx_out[n0 + lane] = (float)code;
        atomicAdd(&hist[code], 1);
    }

    float* qb = quant + (size_t)b * DCH * L2 + t0;
    float lsse = 0.f;
    for (int mm = 0; mm < 16; ++mm) {
        int code = s_idx[mm];
        #pragma unroll
        for (int dd = 0; dd < 2; ++dd) {
            int d   = lane + 32 * dd;
            float q = cb32[code * DCH + d];
            float zv = zb[d * L2 + mm];
            float df = q - zv;
            lsse = fmaf(df, df, lsse);
            qb[d * L2 + mm] = q;
        }
    }
    #pragma unroll
    for (int off = 16; off > 0; off >>= 1)
        lsse += __shfl_xor(lsse, off, 32);
    if (lane == 0) psse[blockIdx.x] = lsse;    // one slot per wave: deterministic
}

// ---------------------------------------------------------------------------
// ConvTranspose1d scalar fallback (only used for dec_t2: CO=1)
// ---------------------------------------------------------------------------
template<int CI, int CO, bool RELU_IN, bool RELU_OUT, bool DENORM>
__global__ void k_tconv(const float* __restrict__ in, const float* __restrict__ w,
                        const float* __restrict__ bias, float* __restrict__ out,
                        int Lin, const float* __restrict__ means,
                        const float* __restrict__ stds) {
    int idx  = blockIdx.x * blockDim.x + threadIdx.x;   // B * CO * (2*Lin)
    int Lout = 2 * Lin;
    int j = idx % Lout;
    int o = (idx / Lout) % CO;
    int b = idx / (Lout * CO);
    int t0 = (j + 1) >> 1;
    int k0 = (j + 1) - 2 * t0;          // 0 or 1
    const float* inb = in + (size_t)b * CI * Lin;
    float acc = bias[o];
    bool ok0 = (t0 < Lin);
    bool ok1 = (t0 >= 1);
    for (int c = 0; c < CI; ++c) {
        const float* wc = w + (c * CO + o) * 4;
        if (ok0) {
            float v = inb[c * Lin + t0];
            if (RELU_IN) v = fmaxf(v, 0.f);
            acc = fmaf(v, wc[k0], acc);
        }
        if (ok1) {
            float v = inb[c * Lin + t0 - 1];
            if (RELU_IN) v = fmaxf(v, 0.f);
            acc = fmaf(v, wc[k0 + 2], acc);
        }
    }
    if (RELU_OUT) acc = fmaxf(acc, 0.f);
    if (DENORM) acc = acc * stds[b] + means[b];
    out[idx] = acc;
}

// ---------------------------------------------------------------------------
// finalize: deterministic loss reduction + perplexity from histogram
// ---------------------------------------------------------------------------
__global__ void k_finalize(const int* __restrict__ hist, const float* __restrict__ psse,
                           float* __restrict__ loss_out, float* __restrict__ perp_out) {
    __shared__ float sh[512];
    int tid = threadIdx.x;
    float s = 0.f;
    for (int i = 0; i < NTOK / 16 / 512; ++i)
        s += psse[tid + 512 * i];
    sh[tid] = s;
    __syncthreads();
    for (int off = 256; off > 0; off >>= 1) {
        if (tid < off) sh[tid] += sh[tid + off];
        __syncthreads();
    }
    float sse = sh[0];
    __syncthreads();
    float p = (float)hist[tid] / (float)NTOK;
    sh[tid] = p * logf(p + 1e-10f);
    __syncthreads();
    for (int off = 256; off > 0; off >>= 1) {
        if (tid < off) sh[tid] += sh[tid + off];
        __syncthreads();
    }
    if (tid == 0) {
        *loss_out = 1.25f * sse / ((float)NTOK * (float)DCH);
        *perp_out = expf(-sh[0]);
    }
}

// ---------------------------------------------------------------------------
// launch
// ---------------------------------------------------------------------------
extern "C" void kernel_launch(void* const* d_in, const int* in_sizes, int n_in,
                              void* d_out, int out_size, void* d_ws, size_t ws_size,
                              hipStream_t stream) {
    const float* x         = (const float*)d_in[0];
    const float* enc_w1    = (const float*)d_in[1];
    const float* enc_b1    = (const float*)d_in[2];
    const float* enc_w2    = (const float*)d_in[3];
    const float* enc_b2    = (const float*)d_in[4];
    const float* enc_w3    = (const float*)d_in[5];
    const float* enc_b3    = (const float*)d_in[6];
    const float* enc_rw1   = (const float*)d_in[7];   // [2,32,128,3]
    const float* enc_rw2   = (const float*)d_in[8];   // [2,128,32,1]
    const float* enc_pre_w = (const float*)d_in[9];
    const float* enc_pre_b = (const float*)d_in[10];
    const float* codebook  = (const float*)d_in[11];
    const float* dec_w1    = (const float*)d_in[12];
    const float* dec_b1    = (const float*)d_in[13];
    const float* dec_rw1   = (const float*)d_in[14];
    const float* dec_rw2   = (const float*)d_in[15];
    const float* dec_t1_w  = (const float*)d_in[16];
    const float* dec_t1_b  = (const float*)d_in[17];
    const float* dec_t2_w  = (const float*)d_in[18];
    const float* dec_t2_b  = (const float*)d_in[19];

    float* W = (float*)d_ws;
    float*    means = W;                      // 64
    float*    stds  = W + 64;                 // 64
    int*      hist  = (int*)(W + 128);        // 512 ints
    float*    cnorm = W + 640;                // 512
    float*    psse  = W + 1152;               // 8192
    _Float16* cb16  = (_Float16*)(W + 9472);  // 32768 halves

    // f16 weight arena (offsets in halves)
    _Float16* wf = (_Float16*)(W + 26112);
    _Float16* w2_16   = wf;                   // 128*256 = 32768
    _Float16* w3_16   = wf + 32768;           // 128*384 = 49152
    _Float16* erw1_16 = wf + 81920;           // 2 * 32*384 = 24576
    _Float16* erw2_16 = wf + 106496;          // 2 * 128*32 = 8192
    _Float16* pre_16  = wf + 114688;          // 64*128    = 8192
    _Float16* dw1_16  = wf + 122880;          // 128*192   = 24576
    _Float16* drw1_16 = wf + 147456;          // 2 * 32*384 = 24576
    _Float16* drw2_16 = wf + 172032;          // 2 * 128*32 = 8192
    _Float16* wt1_16  = wf + 180224;          // 2 * 64*256 = 32768 (end 212992)

    float* bufA = W + 139264;                        // h1 / tconv1 out
    float* bufB = bufA + (size_t)BATCH * 64 * L1;    // h2 / dec trunk
    float* bufC = bufB + (size_t)BATCH * HCH * L2;   // enc trunk
    float* bufR = bufC + (size_t)BATCH * HCH * L2;   // residual hidden
    float* bufZ = bufR + (size_t)BATCH * RCH * L2;   // z
    float* bufQ = bufZ + (size_t)BATCH * DCH * L2;   // quant

    float* out      = (float*)d_out;
    float* out_rec  = out;                   // 524288
    float* out_loss = out + 524288;          // 1
    float* out_idx  = out + 524289;          // 131072
    float* out_perp = out + 655361;          // 1

    k_init<<<2, 256, 0, stream>>>(hist);
    k_revin<<<BATCH, 256, 0, stream>>>(x, means, stds);

    // ---- weight conversions (f32 -> f16) ----
    #define WPREP(src, dst, n) k_wprep<<<((n) + 255) / 256, 256, 0, stream>>>(src, dst, n)
    WPREP(enc_w2, w2_16, 128 * 256);
    WPREP(enc_w3, w3_16, 128 * 384);
    WPREP(enc_rw1,            erw1_16,         32 * 384);
    WPREP(enc_rw1 + 32 * 384, erw1_16 + 12288, 32 * 384);
    WPREP(enc_rw2,            erw2_16,         128 * 32);
    WPREP(enc_rw2 + 128 * 32, erw2_16 + 4096,  128 * 32);
    WPREP(enc_pre_w, pre_16, 64 * 128);
    WPREP(dec_w1,    dw1_16, 128 * 192);
    WPREP(dec_rw1,            drw1_16,         32 * 384);
    WPREP(dec_rw1 + 32 * 384, drw1_16 + 12288, 32 * 384);
    WPREP(dec_rw2,            drw2_16,         128 * 32);
    WPREP(dec_rw2 + 128 * 32, drw2_16 + 4096,  128 * 32);
    #undef WPREP
    k_tw1prep<<<(2 * 64 * 256) / 256, 256, 0, stream>>>(dec_t1_w, wt1_16);
    k_vq_prep<<<2, 256, 0, stream>>>(codebook, cb16, cnorm);

    // ---- encoder ----
    k_enc1<<<(BATCH * 64 * L1) / 256, 256, 0, stream>>>(x, enc_w1, enc_b1, means, stds, bufA);
    // enc2: 64->128, K4, s2, p1, relu-out
    k_conv_wmma<64, 128, 4, 1, 2, 1, false, true, false, true>
        <<<BATCH * (L2 / 16), 256, 0, stream>>>(bufA, w2_16, enc_b2, bufB, L1);
    // enc3: 128->128, K3, p1
    k_conv_wmma<HCH, HCH, 3, 1, 1, 1, false, true, false, false>
        <<<BATCH * (L2 / 16), 256, 0, stream>>>(bufB, w3_16, enc_b3, bufC, L2);
    for (int l = 0; l < 2; ++l) {
        k_conv_wmma<HCH, RCH, 3, 1, 1, 1, true, false, false, false>
            <<<BATCH * (L2 / 64), 256, 0, stream>>>(bufC, erw1_16 + l * 12288, nullptr, bufR, L2);
        k_conv_wmma<RCH, HCH, 1, 0, 1, 1, true, false, true, false>
            <<<BATCH * (L2 / 16), 256, 0, stream>>>(bufR, erw2_16 + l * 4096, nullptr, bufC, L2);
    }
    // final residual relu fused into pre-VQ 1x1 conv
    k_conv_wmma<HCH, DCH, 1, 0, 1, 1, true, true, false, false>
        <<<BATCH * (L2 / 32), 256, 0, stream>>>(bufC, pre_16, enc_pre_b, bufZ, L2);

    // ---- vector quantization (WMMA) ----
    k_vq<<<NTOK / 16, 32, 0, stream>>>(bufZ, cb16, cnorm, codebook, bufQ,
                                       out_idx, hist, psse);

    // ---- decoder ----
    k_conv_wmma<DCH, HCH, 3, 1, 1, 1, false, true, false, false>
        <<<BATCH * (L2 / 16), 256, 0, stream>>>(bufQ, dw1_16, dec_b1, bufB, L2);
    for (int l = 0; l < 2; ++l) {
        k_conv_wmma<HCH, RCH, 3, 1, 1, 1, true, false, false, false>
            <<<BATCH * (L2 / 64), 256, 0, stream>>>(bufB, drw1_16 + l * 12288, nullptr, bufR, L2);
        k_conv_wmma<RCH, HCH, 1, 0, 1, 1, true, false, true, false>
            <<<BATCH * (L2 / 16), 256, 0, stream>>>(bufR, drw2_16 + l * 4096, nullptr, bufB, L2);
    }
    // dec_t1 (ConvTranspose 128->64, s2) as two parity K=2 WMMA convs;
    // residual-stack final relu fused as RELU_IN, output relu fused at store.
    k_conv_wmma<HCH, 64, 2, 1, 1, 2, true, true, false, true>
        <<<BATCH * (L2 / 32), 256, 0, stream>>>(bufB, wt1_16,         dec_t1_b, bufA,     L2);
    k_conv_wmma<HCH, 64, 2, 0, 1, 2, true, true, false, true>
        <<<BATCH * (L2 / 32), 256, 0, stream>>>(bufB, wt1_16 + 16384, dec_t1_b, bufA + 1, L2);
    // dec_t2 (64->1) + RevIN denorm, writes x_recon
    k_tconv<64, 1, false, false, true><<<(BATCH * TLEN) / 256, 256, 0, stream>>>(
        bufA, dec_t2_w, dec_t2_b, out_rec, L1, means, stds);

    // ---- scalar outputs ----
    k_finalize<<<1, 512, 0, stream>>>(hist, psse, out_loss, out_perp);
}
